// LexiconMultiHeadAttention_84902913507683
// MI455X (gfx1250) — matrix operand order repaired and checked by reading, not compiled, not run on hardware
//
#include <hip/hip_runtime.h>
#include <hip/hip_bf16.h>

// ---------------------------------------------------------------------------
// CDNA5 (gfx1250) WMMA types and fragment helpers
// ---------------------------------------------------------------------------
typedef __attribute__((ext_vector_type(16))) __bf16 v16bf;
typedef __attribute__((ext_vector_type(8)))  float  v8f;

union FragBF {
    v16bf v;
    uint4 q[2];
};

// A-matrix fragment, 16x32 bf16 (ISA 7.12.2): lane<16 -> M=lane, K={0..7,16..23};
// lane>=16 -> M=lane-16, K={8..15,24..31}. Two contiguous 16B chunks per lane.
__device__ inline v16bf frag_a16(const __bf16* base, int ld, int lane) {
    const int m  = lane & 15;
    const int ko = (lane < 16) ? 0 : 8;
    const __bf16* p = base + m * ld + ko;
    FragBF f;
    f.q[0] = *(const uint4*)(p);
    f.q[1] = *(const uint4*)(p + 16);
    return f.v;
}

// B-matrix fragment, 32x16 bf16: lane<16 -> N=lane, K={0..7,16..23};
// lane>=16 -> N=lane-16, K={8..15,24..31}. Source is "N-major" (row n = column n).
__device__ inline v16bf frag_b16(const __bf16* base, int ld, int lane) {
    const int n  = lane & 15;
    const int ko = (lane < 16) ? 0 : 8;
    const __bf16* p = base + n * ld + ko;
    FragBF f;
    f.q[0] = *(const uint4*)(p);
    f.q[1] = *(const uint4*)(p + 16);
    return f.v;
}

__device__ inline v8f wmma_bf16(v16bf a, v16bf b, v8f c) {
    return __builtin_amdgcn_wmma_f32_16x16x32_bf16(false, a, false, b, (short)0, c,
                                                   false, false);
}

// ---------------------------------------------------------------------------
// Kernel 1: transpose a 1024x1024 f32 weight matrix into (N,K) bf16
// ---------------------------------------------------------------------------
__global__ void wt_kernel(const float* __restrict__ W, __bf16* __restrict__ Wt) {
    __shared__ float tile[32][33];
    const int n0 = blockIdx.x * 32, k0 = blockIdx.y * 32;
    for (int j = 0; j < 4; j++)
        tile[threadIdx.y + j * 8][threadIdx.x] =
            W[(k0 + threadIdx.y + j * 8) * 1024 + n0 + threadIdx.x];
    __syncthreads();
    for (int j = 0; j < 4; j++)
        Wt[(n0 + threadIdx.y + j * 8) * 1024 + k0 + threadIdx.x] =
            (__bf16)tile[threadIdx.x][threadIdx.y + j * 8];
}

// ---------------------------------------------------------------------------
// Kernel 2: projection GEMM  C = X @ W + b, written as (B,H,L,DK) bf16
//           (or transposed (B,H,DK,L) for V matrices, vtFlag=1)
// block = 128 threads (4 waves), tile = 16 rows x 64 cols, K = 1024
// ---------------------------------------------------------------------------
__global__ void proj_kernel(const float* __restrict__ X, const __bf16* __restrict__ Wt,
                            const float* __restrict__ bias, __bf16* __restrict__ out,
                            int L, int vtFlag) {
    __shared__ __bf16 As[16 * 1024];  // 32 KB: one row-block of X in bf16
    const int tid = threadIdx.x, lane = tid & 31, wave = tid >> 5;
    const int m0 = blockIdx.x * 16;
    for (int idx = tid; idx < 16 * 1024; idx += 128)
        As[idx] = (__bf16)X[m0 * 1024 + idx];
    __syncthreads();

    const int n0 = blockIdx.y * 64 + wave * 16;
    v8f c = {};
    for (int kk = 0; kk < 32; kk++) {
        v16bf a = frag_a16(As + kk * 32, 1024, lane);
        v16bf b = frag_b16(Wt + n0 * 1024 + kk * 32, 1024, lane);
        c = wmma_bf16(a, b, c);
    }
    const int crow = (lane >> 4) * 8, ccol = lane & 15;
    for (int r = 0; r < 8; r++) {
        const int m = m0 + crow + r;
        const int n = n0 + ccol;
        const int b = m / L, l = m % L;
        const int h = n >> 6, dk = n & 63;
        const float val = c[r] + bias[n];
        if (vtFlag)  // (B,H,DK,L)
            out[((b * 16 + h) * 64 + dk) * L + l] = (__bf16)val;
        else         // (B,H,L,DK)
            out[(((b * 16 + h) * L + l) << 6) + dk] = (__bf16)val;
    }
}

// ---------------------------------------------------------------------------
// Kernel 3: fused attention. One block per (b, h, 16 q-rows).
// Scores (16 x LKT f32) and probabilities (16 x LKT bf16) live in LDS.
// Probabilities written to HBM exactly once, with a non-temporal hint so the
// 384 MB stream does not sweep the 192 MB L2 (K/V heads + weights stay hot).
// ---------------------------------------------------------------------------
template <int LKT, int LOG>
__global__ void attn_kernel(const __bf16* __restrict__ qh, const __bf16* __restrict__ kh,
                            const __bf16* __restrict__ vt,
                            const unsigned char* __restrict__ mask,
                            float* __restrict__ attn_out, float* __restrict__ comb,
                            float alpha, int accum) {
    __shared__ float  Ssc[16 * LKT];   // f32 scores / exp values
    __shared__ __bf16 Pbf[16 * LKT];   // bf16 probabilities for P@V WMMA
    __shared__ float  red[128];
    __shared__ float  rmax[16];
    __shared__ float  rinv[16];

    const int tid = threadIdx.x, lane = tid & 31, wave = tid >> 5;
    const int b = blockIdx.z, h = blockIdx.y, q0 = blockIdx.x * 16;
    const int bh = b * 16 + h;
    const int crow = (lane >> 4) * 8, ccol = lane & 15;

    // Q fragments (16 x 64), reused for every key tile
    const __bf16* qbase = qh + (bh * 1024 + q0) * 64;
    const v16bf a0 = frag_a16(qbase, 64, lane);
    const v16bf a1 = frag_a16(qbase + 32, 64, lane);

    // ---- scores: each wave owns LKT/4 key columns -------------------------
    const int cpw = LKT / 4;
    for (int kv0 = wave * cpw; kv0 < (wave + 1) * cpw; kv0 += 16) {
        const __bf16* kb = kh + (bh * LKT + kv0) * 64;
        __builtin_prefetch(kb + 16 * 64, 0, 1);  // global_prefetch_b8 on next tile
        v16bf b0 = frag_b16(kb, 64, lane);
        v16bf b1 = frag_b16(kb + 32, 64, lane);
        v8f c = {};
        c = wmma_bf16(a0, b0, c);
        c = wmma_bf16(a1, b1, c);
        const int col = kv0 + ccol;
        for (int r = 0; r < 8; r++) {
            const int row = crow + r;
            const unsigned char mk =
                __builtin_nontemporal_load(mask + (b * 1024 + q0 + row) * LKT + col);
            Ssc[row * LKT + col] = mk ? -1e9f : c[r] * 0.125f;
        }
    }
    __syncthreads();

    // ---- softmax: 8 threads per row -------------------------------------
    const int r8 = tid >> 3, sg = tid & 7;
    float mx = -3.4e38f;
    for (int col = sg; col < LKT; col += 8) mx = fmaxf(mx, Ssc[r8 * LKT + col]);
    red[tid] = mx;
    __syncthreads();
    if (tid < 16) {
        float m = red[tid * 8];
        for (int j = 1; j < 8; j++) m = fmaxf(m, red[tid * 8 + j]);
        rmax[tid] = m;
    }
    __syncthreads();
    const float rm = rmax[r8];
    float s = 0.f;
    for (int col = sg; col < LKT; col += 8) {
        const float e = __expf(Ssc[r8 * LKT + col] - rm);
        Ssc[r8 * LKT + col] = e;
        s += e;
    }
    red[tid] = s;
    __syncthreads();
    if (tid < 16) {
        float ss = 0.f;
        for (int j = 0; j < 8; j++) ss += red[tid * 8 + j];
        rinv[tid] = 1.0f / ss;
    }
    __syncthreads();

    // ---- normalize + single coalesced NT write of probabilities -----------
    const int obase = bh * 1024 * LKT + q0 * LKT;
    for (int idx = tid; idx < 16 * LKT; idx += 128) {
        const int row = idx >> LOG;
        const float p = Ssc[idx] * rinv[row];
        __builtin_nontemporal_store(p, attn_out + obase + idx);
        Pbf[idx] = (__bf16)p;
    }
    __syncthreads();

    // ---- P @ V: wave w owns dk-tile w ------------------------------------
    const int dk0 = wave * 16;
    v8f acc = {};
    for (int kt = 0; kt < LKT; kt += 32) {
        v16bf a = frag_a16(Pbf + kt, LKT, lane);
        v16bf bb = frag_b16(vt + (bh * 64 + dk0) * LKT + kt, LKT, lane);
        acc = wmma_bf16(a, bb, acc);
    }
    for (int r = 0; r < 8; r++) {
        const int q = q0 + crow + r;
        const int idx = (b * 1024 + q) * 1024 + h * 64 + dk0 + ccol;
        const float v = alpha * acc[r];
        if (accum) comb[idx] += v;
        else       comb[idx] = v;
    }
}

// ---------------------------------------------------------------------------
// Kernel 4: out = comb @ Wo + bo + X, then layernorm. One block per 16 rows,
// 256 threads (8 waves), full 1024-wide row kept in LDS for the LN.
// ---------------------------------------------------------------------------
__global__ void outln_kernel(const float* __restrict__ comb, const __bf16* __restrict__ woT,
                             const float* __restrict__ wob, const float* __restrict__ X,
                             const float* __restrict__ g, const float* __restrict__ be,
                             float* __restrict__ y) {
    __shared__ __bf16 As[16 * 1024];  // 32 KB
    __shared__ float  Yb[16 * 1024];  // 64 KB
    __shared__ float  red[256], red2[256];
    __shared__ float  mu[16], rstd[16];

    const int tid = threadIdx.x, lane = tid & 31, wave = tid >> 5;
    const int m0 = blockIdx.x * 16;
    for (int idx = tid; idx < 16 * 1024; idx += 256)
        As[idx] = (__bf16)comb[m0 * 1024 + idx];
    __syncthreads();

    const int crow = (lane >> 4) * 8, ccol = lane & 15;
    for (int nt = 0; nt < 8; nt++) {
        const int n0 = wave * 128 + nt * 16;
        v8f c = {};
        for (int kk = 0; kk < 32; kk++) {
            v16bf a = frag_a16(As + kk * 32, 1024, lane);
            v16bf b = frag_b16(woT + n0 * 1024 + kk * 32, 1024, lane);
            c = wmma_bf16(a, b, c);
        }
        for (int r = 0; r < 8; r++) {
            const int rr = crow + r, n = n0 + ccol;
            Yb[rr * 1024 + n] = c[r] + wob[n];
        }
    }
    __syncthreads();

    const int rr = tid >> 4, tt = tid & 15;
    float s = 0.f, s2 = 0.f;
    for (int i = 0; i < 64; i++) {
        const int col = tt + i * 16;
        const float v = Yb[rr * 1024 + col] + X[(m0 + rr) * 1024 + col];
        Yb[rr * 1024 + col] = v;
        s += v;
        s2 += v * v;
    }
    red[tid] = s;
    red2[tid] = s2;
    __syncthreads();
    if (tt == 0) {
        float ss = 0.f, ss2 = 0.f;
        for (int j = 0; j < 16; j++) { ss += red[rr * 16 + j]; ss2 += red2[rr * 16 + j]; }
        const float m = ss * (1.0f / 1024.0f);
        const float var = ss2 * (1.0f / 1024.0f) - m * m;
        mu[rr] = m;
        rstd[rr] = rsqrtf(var + 1e-5f);
    }
    __syncthreads();
    for (int i = 0; i < 64; i++) {
        const int col = tt + i * 16;
        const float v = (Yb[rr * 1024 + col] - mu[rr]) * rstd[rr] * g[col] + be[col];
        __builtin_nontemporal_store(v, y + (m0 + rr) * 1024 + col);
    }
}

// ---------------------------------------------------------------------------
// Host launcher
// ---------------------------------------------------------------------------
extern "C" void kernel_launch(void* const* d_in, const int* in_sizes, int n_in,
                              void* d_out, int out_size, void* d_ws, size_t ws_size,
                              hipStream_t stream) {
    (void)in_sizes; (void)n_in; (void)out_size; (void)ws_size;
    const float* q   = (const float*)d_in[0];
    const float* k   = (const float*)d_in[1];
    const float* v   = (const float*)d_in[2];
    const float* kl  = (const float*)d_in[3];
    const float* vl  = (const float*)d_in[4];
    const unsigned char* pm  = (const unsigned char*)d_in[5];
    const unsigned char* pml = (const unsigned char*)d_in[6];
    const float* wq_w  = (const float*)d_in[7];
    const float* wq_b  = (const float*)d_in[8];
    const float* wk_w  = (const float*)d_in[9];
    const float* wk_b  = (const float*)d_in[10];
    const float* wv_w  = (const float*)d_in[11];
    const float* wv_b  = (const float*)d_in[12];
    const float* wkl_w = (const float*)d_in[13];
    const float* wkl_b = (const float*)d_in[14];
    const float* wvl_w = (const float*)d_in[15];
    const float* wvl_b = (const float*)d_in[16];
    const float* wo_w  = (const float*)d_in[17];
    const float* wo_b  = (const float*)d_in[18];
    const float* ln_g  = (const float*)d_in[19];
    const float* ln_b  = (const float*)d_in[20];

    char* ws = (char*)d_ws;
    const size_t MB = 1024 * 1024;
    __bf16* wqT  = (__bf16*)(ws + 0 * MB);
    __bf16* wkT  = (__bf16*)(ws + 2 * MB);
    __bf16* wvT  = (__bf16*)(ws + 4 * MB);
    __bf16* wklT = (__bf16*)(ws + 6 * MB);
    __bf16* wvlT = (__bf16*)(ws + 8 * MB);
    __bf16* woT  = (__bf16*)(ws + 10 * MB);
    __bf16* qh   = (__bf16*)(ws + 12 * MB);  // (B,H,LQ,DK)
    __bf16* khh  = (__bf16*)(ws + 20 * MB);  // (B,H,LK,DK)
    __bf16* vtt  = (__bf16*)(ws + 28 * MB);  // (B,H,DK,LK)
    __bf16* klh  = (__bf16*)(ws + 36 * MB);  // (B,H,LKL,DK)
    __bf16* vlt  = (__bf16*)(ws + 40 * MB);  // (B,H,DK,LKL)
    float*  comb = (float*)(ws + 44 * MB);   // (B,LQ,H*DK) f32

    float* y_out  = (float*)d_out;
    float* attn_c = y_out + 4 * 1024 * 1024;
    float* attn_l = attn_c + 4 * 16 * 1024 * 1024;

    // 1) weight transpose + bf16 convert
    dim3 tb(32, 8), tg(32, 32);
    wt_kernel<<<tg, tb, 0, stream>>>(wq_w, wqT);
    wt_kernel<<<tg, tb, 0, stream>>>(wk_w, wkT);
    wt_kernel<<<tg, tb, 0, stream>>>(wv_w, wvT);
    wt_kernel<<<tg, tb, 0, stream>>>(wkl_w, wklT);
    wt_kernel<<<tg, tb, 0, stream>>>(wvl_w, wvlT);
    wt_kernel<<<tg, tb, 0, stream>>>(wo_w, woT);

    // 2) projections (WMMA bf16)
    proj_kernel<<<dim3(256, 16), 128, 0, stream>>>(q, wqT, wq_b, qh, 1024, 0);
    proj_kernel<<<dim3(256, 16), 128, 0, stream>>>(k, wkT, wk_b, khh, 1024, 0);
    proj_kernel<<<dim3(256, 16), 128, 0, stream>>>(v, wvT, wv_b, vtt, 1024, 1);
    proj_kernel<<<dim3(128, 16), 128, 0, stream>>>(kl, wklT, wkl_b, klh, 512, 0);
    proj_kernel<<<dim3(128, 16), 128, 0, stream>>>(vl, wvlT, wvl_b, vlt, 512, 1);

    // 3) fused attention: context writes comb, lexicon accumulates
    attn_kernel<1024, 10><<<dim3(64, 16, 4), 128, 0, stream>>>(
        qh, khh, vtt, pm, attn_c, comb, 0.5f, 0);
    attn_kernel<512, 9><<<dim3(64, 16, 4), 128, 0, stream>>>(
        qh, klh, vlt, pml, attn_l, comb, 0.5f, 1);

    // 4) output projection + residual + layernorm
    outln_kernel<<<256, 256, 0, stream>>>(comb, woT, wo_b, q, ln_g, ln_b, y_out);
}
